// MultiHeadAttention_36386962932540
// MI455X (gfx1250) — matrix operand order repaired and checked by reading, not compiled, not run on hardware
//
#include <hip/hip_runtime.h>

// ---------------------------------------------------------------------------
// Types for CDNA5 WMMA (wave32): v_wmma_f32_16x16x32_bf16
// ---------------------------------------------------------------------------
typedef __attribute__((ext_vector_type(16))) __bf16        v16bf;
typedef __attribute__((ext_vector_type(8)))  float         v8f;
typedef __attribute__((ext_vector_type(4)))  unsigned int  v4u;
typedef __attribute__((ext_vector_type(8)))  unsigned int  v8u;

#define NB 2          // batch
#define NS 2048       // sequence
#define ND 1024       // model dim
#define NH 16         // heads
#define HD 64         // head dim (dqk == dv)

__device__ __forceinline__ unsigned short f2bf(float f) {
    unsigned int u = __builtin_bit_cast(unsigned int, f);
    unsigned int r = u + 0x7FFFu + ((u >> 16) & 1u);   // round-to-nearest-even
    return (unsigned short)(r >> 16);
}

__device__ __forceinline__ v16bf frag_from(const unsigned short* p0,
                                           const unsigned short* p1) {
    v8u r;
    r.lo = *(const v4u*)p0;
    r.hi = *(const v4u*)p1;
    return __builtin_bit_cast(v16bf, r);
}

// A fragment: 16x32 bf16 tile, row-major source, leading dim lda (elements).
__device__ __forceinline__ v16bf load_a(const unsigned short* a, int lda, int lane) {
    const int half = (lane >> 4) & 1, m = lane & 15;
    const unsigned short* p = a + (size_t)m * lda + half * 8;
    return frag_from(p, p + 16);   // K = {half*8..+7} and {16+half*8..+7}
}

// B fragment: 32x16 (KxN), source stored column-major bT[n][k], leading dim ldb.
__device__ __forceinline__ v16bf load_bT(const unsigned short* bT, int ldb, int lane) {
    const int half = (lane >> 4) & 1, n = lane & 15;
    const unsigned short* p = bT + (size_t)n * ldb + half * 8;
    return frag_from(p, p + 16);
}

__device__ __forceinline__ v8f wmma_bf16(v16bf a, v16bf b, v8f c) {
    return __builtin_amdgcn_wmma_f32_16x16x32_bf16(
        false, a, false, b, (short)0, c, false, false);
}

// Async global -> LDS staging of one 16-byte packet (VGLOBAL op 98, ASYNCcnt).
__device__ __forceinline__ void async_g2l_b128(unsigned ldsOff,
                                               const unsigned short* gaddr) {
    asm volatile("global_load_async_to_lds_b128 %0, %1, off"
                 :: "v"(ldsOff), "v"(gaddr) : "memory");
}

// ---------------------------------------------------------------------------
// Kernel 1a: fp32 -> bf16 copy
// ---------------------------------------------------------------------------
__global__ void cvt_bf16_k(const float* __restrict__ src,
                           unsigned short* __restrict__ dst, int n) {
    int i = blockIdx.x * 256 + threadIdx.x;
    if (i < n) dst[i] = f2bf(src[i]);
}

// Kernel 1b: fp32 W[K][N] -> bf16 Wt[N][K] (column-major for B-fragments)
__global__ void transpose_bf16_k(const float* __restrict__ W,
                                 unsigned short* __restrict__ Wt, int K, int N) {
    int i = blockIdx.x * 256 + threadIdx.x;
    if (i < K * N) {
        int k = i / N, n = i % N;
        Wt[(size_t)n * K + k] = f2bf(W[i]);
    }
}

// ---------------------------------------------------------------------------
// Kernel 2: projection GEMM  out[4096,1024] = X[4096,1024] @ W  (bf16 wmma)
//   mode 0: Q (scaled 1/8), mode 1: K, mode 2: V stored head-transposed.
// Block = 128 thr (4 waves); wave = 16 rows x 64 cols.
// k-loop unrolled by 64 with ping-pong A/B register buffers so the loads of
// step k+32 are in flight while the 4 WMMAs of step k execute.
// ---------------------------------------------------------------------------
__global__ void gemm_proj_k(const unsigned short* __restrict__ X,
                            const unsigned short* __restrict__ Wt,
                            unsigned short* __restrict__ outQK,
                            unsigned short* __restrict__ outVt,
                            int mode, float scale) {
    const int lane = threadIdx.x & 31;
    const int wave = threadIdx.x >> 5;
    const int r0 = blockIdx.x * 64 + wave * 16;
    const int c0 = blockIdx.y * 64;

    const unsigned short* Abase = X + (size_t)r0 * ND;

    v8f acc[4] = {};
    v16bf aA = load_a(Abase, ND, lane);
    v16bf bA[4];
#pragma unroll
    for (int t = 0; t < 4; ++t)
        bA[t] = load_bT(Wt + (size_t)(c0 + t * 16) * ND, ND, lane);

    for (int k = 0; k < ND; k += 64) {
        v16bf aB = load_a(Abase + k + 32, ND, lane);
        v16bf bB[4];
#pragma unroll
        for (int t = 0; t < 4; ++t)
            bB[t] = load_bT(Wt + (size_t)(c0 + t * 16) * ND + k + 32, ND, lane);
#pragma unroll
        for (int t = 0; t < 4; ++t) acc[t] = wmma_bf16(aA, bA[t], acc[t]);
        if (k + 64 < ND) {
            aA = load_a(Abase + k + 64, ND, lane);
#pragma unroll
            for (int t = 0; t < 4; ++t)
                bA[t] = load_bT(Wt + (size_t)(c0 + t * 16) * ND + k + 64, ND, lane);
        }
#pragma unroll
        for (int t = 0; t < 4; ++t) acc[t] = wmma_bf16(aB, bB[t], acc[t]);
    }

    const int half = lane >> 4, n = lane & 15;
#pragma unroll
    for (int t = 0; t < 4; ++t) {
#pragma unroll
        for (int i = 0; i < 8; ++i) {
            int row = r0 + i + 8 * half;           // C layout: M = i + 8*(l>>4)
            int col = c0 + t * 16 + n;             //           N = l & 15
            unsigned short bv = f2bf(acc[t][i] * scale);
            if (mode == 2) {                        // Vt[(b*NH+h)*HD + dv][s]
                int b_ = row >> 11, s = row & (NS - 1);
                int h = col >> 6, dv = col & (HD - 1);
                outVt[((size_t)((b_ * NH + h) * HD + dv) << 11) + s] = bv;
            } else {
                outQK[(size_t)row * ND + col] = bv;
            }
        }
    }
}

// ---------------------------------------------------------------------------
// Kernel 3: attention  O = relu15(Qh @ Kh^T) @ V  per (b,h)
// Block = 128 thr (4 waves). WG = 64 queries; wave = 16 queries x 64 dv.
// V tiles (64 dv x 32 keys, 4KB) are staged in LDS via async global->LDS
// (double-buffered, s_wait_asynccnt + one barrier per step) and shared by the
// 4 waves. K rows are prefetched one cacheline/lane one step ahead.
// Score fragments re-staged through per-wave LDS (ReLU15 + bf16 fused).
// ---------------------------------------------------------------------------
__global__ void attn_relu15_k(const unsigned short* __restrict__ Qh,
                              const unsigned short* __restrict__ Kh,
                              const unsigned short* __restrict__ Vt,
                              unsigned short* __restrict__ Obf) {
    __shared__ __align__(16) unsigned short sV[2][HD * 32];   // 2 x 4KB V tiles
    __shared__ __align__(16) unsigned short sS[4][16 * 32];   // 1KB per wave

    const int lane = threadIdx.x & 31;
    const int wave = threadIdx.x >> 5;
    const int bh = blockIdx.y;                  // b*NH + h
    const int b = bh >> 4, h = bh & (NH - 1);
    const int q0 = blockIdx.x * 64 + wave * 16;

    const unsigned short* Qbase = Qh + (size_t)(b * NS + q0) * ND + h * HD;
    const unsigned short* Kbase = Kh + (size_t)(b * NS) * ND + h * HD;
    const unsigned short* Vbase = Vt + ((size_t)(bh * HD) << 11);  // [64][2048]

    v16bf aq0 = load_a(Qbase, ND, lane);        // dqk 0..31 (already /temp)
    v16bf aq1 = load_a(Qbase + 32, ND, lane);   // dqk 32..63
    v8f o[4] = {};

    const int half = lane >> 4, n = lane & 15;
    unsigned short* myS = sS[wave];

    // Prologue: stage V tile for kb=0 into buffer 0 (256 x b128, 2 per thread)
    {
        int idx = threadIdx.x;
#pragma unroll
        for (int r = 0; r < 2; ++r, idx += 128) {
            int row = idx >> 2, part = idx & 3;
            unsigned lo = (unsigned)(size_t)(const void*)&sV[0][row * 32 + part * 8];
            async_g2l_b128(lo, Vbase + (size_t)row * NS + part * 8);
        }
    }

    for (int kb = 0; kb < NS; kb += 32) {
        const int buf = (kb >> 5) & 1;
        if (kb + 32 < NS)   // prefetch next K tile: one cacheline per lane
            __builtin_prefetch(Kbase + (size_t)(kb + 32 + lane) * ND, 0, 3);

#pragma unroll
        for (int c = 0; c < 2; ++c) {           // two 16-key score chunks
            const unsigned short* kp = Kbase + (size_t)(kb + c * 16) * ND;
            v16bf b0 = load_bT(kp, ND, lane);        // Kt columns = K rows
            v16bf b1 = load_bT(kp + 32, ND, lane);
            v8f s = {};
            s = wmma_bf16(aq0, b0, s);
            s = wmma_bf16(aq1, b1, s);
#pragma unroll
            for (int i = 0; i < 8; ++i) {       // relu15 + bf16 -> LDS tile
                float x = s[i];
                x = fminf(fmaxf(x, 0.0f), 15.0f);
                myS[(i + 8 * half) * 32 + c * 16 + n] = f2bf(x);
            }
        }
        asm volatile("s_wait_dscnt 0" ::: "memory");       // sS RAW (same wave)
        asm volatile("s_wait_asynccnt 0x0" ::: "memory");  // our V-tile packets
        __syncthreads();                                   // sV[buf] valid WG-wide

        if (kb + 32 < NS) {   // stage next V tile into the other buffer
            int idx = threadIdx.x;
#pragma unroll
            for (int r = 0; r < 2; ++r, idx += 128) {
                int row = idx >> 2, part = idx & 3;
                unsigned lo =
                    (unsigned)(size_t)(const void*)&sV[buf ^ 1][row * 32 + part * 8];
                async_g2l_b128(lo, Vbase + (size_t)row * NS + (kb + 32) + part * 8);
            }
        }

        v16bf a2 = load_a(myS, 32, lane);       // scores as A fragment
#pragma unroll
        for (int t = 0; t < 4; ++t) {
            v16bf bv = load_bT(&sV[buf][(t * 16) * 32], 32, lane);
            o[t] = wmma_bf16(a2, bv, o[t]);
        }
    }

#pragma unroll
    for (int t = 0; t < 4; ++t) {
#pragma unroll
        for (int i = 0; i < 8; ++i) {
            int row = q0 + i + 8 * half;
            int col = h * HD + t * 16 + n;
            Obf[(size_t)(b * NS + row) * ND + col] = f2bf(o[t][i]);
        }
    }
}

// ---------------------------------------------------------------------------
// Kernel 4: out = LayerNorm( Obf @ Wfc + residual ) * gamma + beta
// Block = 256 thr (8 waves). WG = 16 rows x 1024 cols; wave = 16 x 128.
// Same ping-pong register pipelining as kernel 2; post-GEMM tile staged in
// 64KB LDS for the row reductions.
// ---------------------------------------------------------------------------
__global__ void fc_ln_k(const unsigned short* __restrict__ Obf,
                        const unsigned short* __restrict__ Wt,
                        const float* __restrict__ resid,
                        const float* __restrict__ gamma,
                        const float* __restrict__ beta,
                        float* __restrict__ out) {
    __shared__ float xt[16 * ND];               // 64 KB (<< 320 KB/WGP)

    const int lane = threadIdx.x & 31;
    const int wave = threadIdx.x >> 5;
    const int r0 = blockIdx.x * 16;
    const int c0 = wave * 128;

    const unsigned short* Abase = Obf + (size_t)r0 * ND;

    v8f acc[8] = {};
    v16bf aA = load_a(Abase, ND, lane);
    v16bf bA[8];
#pragma unroll
    for (int t = 0; t < 8; ++t)
        bA[t] = load_bT(Wt + (size_t)(c0 + t * 16) * ND, ND, lane);

    for (int k = 0; k < ND; k += 64) {
        v16bf aB = load_a(Abase + k + 32, ND, lane);
        v16bf bB[8];
#pragma unroll
        for (int t = 0; t < 8; ++t)
            bB[t] = load_bT(Wt + (size_t)(c0 + t * 16) * ND + k + 32, ND, lane);
#pragma unroll
        for (int t = 0; t < 8; ++t) acc[t] = wmma_bf16(aA, bA[t], acc[t]);
        if (k + 64 < ND) {
            aA = load_a(Abase + k + 64, ND, lane);
#pragma unroll
            for (int t = 0; t < 8; ++t)
                bA[t] = load_bT(Wt + (size_t)(c0 + t * 16) * ND + k + 64, ND, lane);
        }
#pragma unroll
        for (int t = 0; t < 8; ++t) acc[t] = wmma_bf16(aB, bB[t], acc[t]);
    }

    const int half = lane >> 4, n = lane & 15;
#pragma unroll
    for (int t = 0; t < 8; ++t) {
#pragma unroll
        for (int i = 0; i < 8; ++i) {
            int row = i + 8 * half;             // local 0..15
            int col = c0 + t * 16 + n;
            xt[row * ND + col] =
                acc[t][i] + resid[(size_t)(r0 + row) * ND + col];
        }
    }
    __syncthreads();

    const int r = threadIdx.x >> 4;             // 0..15 (row)
    const int j = threadIdx.x & 15;             // 16 lanes per row
    float s = 0.f, ss = 0.f;
    for (int c = j; c < ND; c += 16) {
        float x = xt[r * ND + c];
        s += x; ss += x * x;
    }
#pragma unroll
    for (int off = 8; off > 0; off >>= 1) {     // reduce within 16-lane group
        s  += __shfl_xor(s, off, 16);
        ss += __shfl_xor(ss, off, 16);
    }
    const float mu = s * (1.0f / ND);
    const float var = ss * (1.0f / ND) - mu * mu;
    const float rs = rsqrtf(var + 1e-6f);
    for (int c = j; c < ND; c += 16) {
        float x = xt[r * ND + c];
        out[(size_t)(r0 + r) * ND + c] = (x - mu) * rs * gamma[c] + beta[c];
    }
}

// ---------------------------------------------------------------------------
// Host launch
// ---------------------------------------------------------------------------
extern "C" void kernel_launch(void* const* d_in, const int* in_sizes, int n_in,
                              void* d_out, int out_size, void* d_ws, size_t ws_size,
                              hipStream_t stream) {
    const float* q    = (const float*)d_in[0];
    const float* k    = (const float*)d_in[1];
    const float* v    = (const float*)d_in[2];
    const float* Wq   = (const float*)d_in[3];
    const float* Wk   = (const float*)d_in[4];
    const float* Wv   = (const float*)d_in[5];
    const float* Wfc  = (const float*)d_in[6];
    const float* gamma = (const float*)d_in[7];
    const float* beta  = (const float*)d_in[8];
    float* out = (float*)d_out;

    char* ws = (char*)d_ws;
    const size_t MB = 1024 * 1024;
    // Workspace layout (56 MB total, Obf aliases qb after projections):
    unsigned short* Wqt  = (unsigned short*)(ws + 0 * MB);   // 2 MB
    unsigned short* Wkt  = (unsigned short*)(ws + 2 * MB);   // 2 MB
    unsigned short* Wvt  = (unsigned short*)(ws + 4 * MB);   // 2 MB
    unsigned short* Wfct = (unsigned short*)(ws + 6 * MB);   // 2 MB
    unsigned short* qb   = (unsigned short*)(ws + 8 * MB);   // 8 MB
    unsigned short* kb   = (unsigned short*)(ws + 16 * MB);  // 8 MB
    unsigned short* vb   = (unsigned short*)(ws + 24 * MB);  // 8 MB
    unsigned short* QhB  = (unsigned short*)(ws + 32 * MB);  // 8 MB (pre-scaled)
    unsigned short* KhB  = (unsigned short*)(ws + 40 * MB);  // 8 MB
    unsigned short* VtB  = (unsigned short*)(ws + 48 * MB);  // 8 MB
    unsigned short* Obf  = qb;  // qb dead after Q projection

    const int nTok = NB * NS * ND;      // 4,194,304
    const int nW   = ND * ND;           // 1,048,576

    // 1) bf16 conversions + weight transposes
    cvt_bf16_k<<<(nTok + 255) / 256, 256, 0, stream>>>(q, qb, nTok);
    cvt_bf16_k<<<(nTok + 255) / 256, 256, 0, stream>>>(k, kb, nTok);
    cvt_bf16_k<<<(nTok + 255) / 256, 256, 0, stream>>>(v, vb, nTok);
    transpose_bf16_k<<<(nW + 255) / 256, 256, 0, stream>>>(Wq,  Wqt,  ND, ND);
    transpose_bf16_k<<<(nW + 255) / 256, 256, 0, stream>>>(Wk,  Wkt,  ND, ND);
    transpose_bf16_k<<<(nW + 255) / 256, 256, 0, stream>>>(Wv,  Wvt,  ND, ND);
    transpose_bf16_k<<<(nW + 255) / 256, 256, 0, stream>>>(Wfc, Wfct, ND, ND);

    // 2) QKV projections (Q pre-scaled by 1/sqrt(dqk) = 0.125)
    dim3 gGemm(NB * NS / 64, ND / 64);          // (64, 16)
    gemm_proj_k<<<gGemm, 128, 0, stream>>>(qb, Wqt, QhB, nullptr, 0, 0.125f);
    gemm_proj_k<<<gGemm, 128, 0, stream>>>(kb, Wkt, KhB, nullptr, 1, 1.0f);
    gemm_proj_k<<<gGemm, 128, 0, stream>>>(vb, Wvt, nullptr, VtB, 2, 1.0f);

    // 3) attention: relu15(QK^T) @ V
    dim3 gAttn(NS / 64, NB * NH);               // (32, 32)
    attn_relu15_k<<<gAttn, 128, 0, stream>>>(QhB, KhB, VtB, Obf);

    // 4) FC + residual + LayerNorm
    fc_ln_k<<<NB * NS / 16, 256, 0, stream>>>(Obf, Wfct, q, gamma, beta, out);
}
